// Dropout_Conv2D_41266045780319
// MI455X (gfx1250) — matrix operand order repaired and checked by reading, not compiled, not run on hardware
//
#include <hip/hip_runtime.h>
#include <hip/hip_bf16.h>

typedef __attribute__((ext_vector_type(8)))  float    v8f;
typedef __attribute__((ext_vector_type(4)))  int      v4i;
typedef __attribute__((ext_vector_type(8)))  int      v8i;
typedef __attribute__((ext_vector_type(4)))  unsigned v4u;
typedef __attribute__((ext_vector_type(4)))  __bf16   v4bf;
typedef __attribute__((ext_vector_type(8)))  __bf16   v8bf;
typedef __attribute__((ext_vector_type(16))) __bf16   v16bf;

#define B_     32
#define CIN    128
#define COUT   256
#define NTAP   9      /* 3x3 */
#define KTOT   1152   /* CIN * 9 */
#define NPIX   4096   /* 64 * 64 */
#define PDROP  0.2f

#define BM 64
#define BN 128
#define BK 32
#define NITER 36      /* 9 taps * (CIN/BK) chunks */

#define AS1 __attribute__((address_space(1)))
#define AS3 __attribute__((address_space(3)))

#if defined(__AMDGCN__) && __has_builtin(__builtin_amdgcn_tensor_load_to_lds)
#define USE_TDM 1
#else
#define USE_TDM 0
#endif

#if defined(__AMDGCN__) && __has_builtin(__builtin_amdgcn_global_load_async_to_lds_b128)
#define USE_ASYNC 1
#else
#define USE_ASYNC 0
#endif

#if USE_ASYNC
# if __has_builtin(__builtin_amdgcn_s_wait_asynccnt)
#  define ASYNC_WAIT() __builtin_amdgcn_s_wait_asynccnt(0)
# else
#  define ASYNC_WAIT() asm volatile("s_wait_asynccnt 0x0" ::: "memory")
# endif
#else
# define ASYNC_WAIT() ((void)0)
#endif

#if USE_TDM
# if __has_builtin(__builtin_amdgcn_s_wait_tensorcnt)
#  define TENSOR_WAIT() __builtin_amdgcn_s_wait_tensorcnt(0)
# else
#  define TENSOR_WAIT() asm volatile("s_wait_tensorcnt 0x0" ::: "memory")
# endif
#endif

// ---------------------------------------------------------------------------
// Kernel 1: weight-dropout mask + fp32->bf16, repacked [B][tap][O][Cin]
// so every GEMM A-tile is a dense 2D tile (rows stride CIN) — TDM-friendly.
// ---------------------------------------------------------------------------
__global__ __launch_bounds__(256) void prep_weights(
    const float* __restrict__ w, const float* __restrict__ uw,
    __bf16* __restrict__ wb)
{
    size_t j = ((size_t)blockIdx.x * 256 + threadIdx.x) * 4;  // flat [B][9][O][CIN]
    int    ci  = (int)(j & (CIN - 1));
    size_t r   = j >> 7;                    // / CIN
    int    o   = (int)(r & (COUT - 1));
    size_t r2  = r >> 8;                    // / COUT
    int    tap = (int)(r2 % NTAP);
    int    b   = (int)(r2 / NTAP);

    v4bf out;
#pragma unroll
    for (int e = 0; e < 4; ++e) {
        size_t si = ((size_t)o * CIN + (ci + e)) * NTAP + tap;                 // weight[o][ci][tap]
        size_t ui = ((size_t)b * COUT + o) * KTOT + (size_t)(ci + e) * NTAP + tap;
        out[e] = (__bf16)(uw[ui] > PDROP ? w[si] : 0.0f);
    }
    *(v4bf*)(wb + j) = out;
}

// ---------------------------------------------------------------------------
// Kernel 2: per-tap implicit-GEMM conv, v_wmma_f32_16x16x32_bf16.
// A tile: TDM tensor_load_to_lds (1 descriptor/iter, wave 0) -> fallback
//         async global->LDS b128 -> fallback sync copy. Double-buffered LDS.
//   y[o,p] = sum_{tap} sum_{ci} W[b,tap,o,ci] * x[b,ci, p + off(tap)]
// ---------------------------------------------------------------------------
__global__ __launch_bounds__(256) void conv_wmma(
    const float*  __restrict__ x,
    const __bf16* __restrict__ wb,
    const float*  __restrict__ bias,
    const float*  __restrict__ ub,
    float*        __restrict__ y)
{
    __shared__ __align__(64) __bf16 lds_a[2][BM * BK];  // [m][k], k = ci chunk
    __shared__ __align__(64) __bf16 lds_b[2][BN * BK];  // [n][k]

    const int tid  = threadIdx.x;
    const int wave = tid >> 5;
    const int lane = tid & 31;
    const int sel  = lane >> 4;
    const int l15  = lane & 15;

    const int b  = blockIdx.z;
    const int M0 = blockIdx.y * BM;
    const int N0 = blockIdx.x * BN;

    const int mw = wave & 3;     // 16-row strip
    const int nw = wave >> 2;    // 64-col half

    v8f acc0 = {}, acc1 = {}, acc2 = {}, acc3 = {};

    // A-tile fallback fill mapping: one 16B transfer per thread
    const int am = tid >> 2;               // 0..63 (O row in tile)
    const int ak = (tid & 3) * 8;          // 0,8,16,24 (ci in chunk)
    const int aRow = M0 + am;
    const __bf16* wbb = wb + (size_t)b * NTAP * COUT * CIN;

#if USE_TDM
    // tile base (row M0) of the [tap][O][CIN] weight block for this sample
    const unsigned long long wtile_base =
        (unsigned long long)(uintptr_t)(wbb + (size_t)M0 * CIN);
    const unsigned ldsA0 = (unsigned)(uintptr_t)(AS3 char*)&lds_a[0][0];
    const unsigned ldsA1 = (unsigned)(uintptr_t)(AS3 char*)&lds_a[1][0];
#endif

    // B-tile fill mapping: 16 channels for one pixel per thread
    const int bn = tid >> 1;               // 0..127 pixel in tile
    const int ks = (tid & 1) * 16;         // ci offset 0 or 16
    const int p  = N0 + bn;
    const int py = p >> 6;
    const int px = p & 63;
    const float* xb = x + (size_t)b * CIN * NPIX;

    auto fill_a = [&](int buf, int kk) {
        const int tap = kk >> 2;           // 0..8
        const int c0  = (kk & 3) * BK;     // ci chunk base
#if USE_TDM
        if (wave == 0) {
            // D# per CDNA5 ISA: 2D tile, data_size=2B,
            // tile 32(x) x 64(y), row stride CIN elements.
            unsigned long long ga =
                wtile_base + 2ull * ((size_t)tap * (COUT * CIN) + c0);
            v4u g0;
            g0.x = 1u;                                   // count=1 (valid D#)
            g0.y = (buf ? ldsA1 : ldsA0);                // lds_addr (bytes)
            g0.z = (unsigned)ga;                         // global_addr[31:0]
            g0.w = (unsigned)(ga >> 32) | 0x80000000u;   // addr[56:32] | type=2
            v8i g1;
            g1[0] = 0x10000;                             // data_size=1 (2 bytes)
            g1[1] = (CIN  & 0xffff) << 16;               // tensor_dim0 lo
            g1[2] = (CIN >> 16) | ((COUT & 0xffff) << 16); // dim0 hi | dim1 lo
            g1[3] = (COUT >> 16) | (BK << 16);           // dim1 hi | tile_dim0=32
            g1[4] = BM;                                  // tile_dim1=64, tile_dim2=0
            g1[5] = CIN;                                 // tensor_dim0_stride lo
            g1[6] = 0;                                   // stride0 hi | stride1 lo
            g1[7] = 0;                                   // stride1 hi
            v4i gz4 = {0, 0, 0, 0};                      // groups 2/3 unused (2D)
            v8i gz8 = {0, 0, 0, 0, 0, 0, 0, 0};          // extra group (clang-23 form)
            __builtin_amdgcn_tensor_load_to_lds(g0, g1, gz4, gz4, gz8, 0);
        }
#else
        const __bf16* asrc = wbb + ((size_t)tap * COUT + aRow) * CIN + c0 + ak;
        __bf16* adst = &lds_a[buf][am * BK + ak];
# if USE_ASYNC
        __builtin_amdgcn_global_load_async_to_lds_b128(
            (AS1 v4i*)asrc, (AS3 v4i*)adst, 0, 0);
# else
        *(v8bf*)adst = *(const v8bf*)asrc;
# endif
#endif
    };

    auto fill_b = [&](int buf, int kk) {
        const int tap = kk >> 2;
        const int c0  = (kk & 3) * BK;
        const int kh = tap / 3;
        const int kw = tap - kh * 3;
        const int h  = py + kh - 1;
        const int w  = px + kw - 1;
        const bool valid = ((unsigned)h < 64u) && ((unsigned)w < 64u);
        const float* bsrc = xb + ((size_t)(c0 + ks) << 12) + (h << 6) + w;
        v4bf tmp[4];
#pragma unroll
        for (int q = 0; q < 4; ++q)
#pragma unroll
            for (int e = 0; e < 4; ++e) {
                float v = valid ? bsrc[(size_t)(q * 4 + e) << 12] : 0.0f;
                tmp[q][e] = (__bf16)v;
            }
#pragma unroll
        for (int q = 0; q < 4; ++q)
            *(v4bf*)&lds_b[buf][bn * BK + ks + q * 4] = tmp[q];
    };

    fill_a(0, 0);
    fill_b(0, 0);
#if USE_TDM
    if (wave == 0) TENSOR_WAIT();
#else
    ASYNC_WAIT();
#endif
    __syncthreads();

    for (int kk = 0; kk < NITER; ++kk) {
        const int cur = kk & 1;
        if (kk + 1 < NITER) {                 // overlap next-tile fill with WMMAs
            fill_a(cur ^ 1, kk + 1);
            fill_b(cur ^ 1, kk + 1);
        }

        // A fragment (16x32 bf16, contiguous k: lanes 0-15 K0..15, hi K16..31)
        const __bf16* arow = &lds_a[cur][(mw * 16 + l15) * BK];
        v8bf alo = *(const v8bf*)(arow + sel * 16);
        v8bf ahi = *(const v8bf*)(arow + sel * 16 + 8);
        v16bf afrag = __builtin_shufflevector(alo, ahi,
            0, 1, 2, 3, 4, 5, 6, 7, 8, 9, 10, 11, 12, 13, 14, 15);

        // B fragments: [n][k] layout -> 32 contiguous bytes per lane
        const int nb = (nw * 64 + l15) * BK + sel * 16;
        v16bf b0 = *(const v16bf*)&lds_b[cur][nb + 0 * 16 * BK];
        v16bf b1 = *(const v16bf*)&lds_b[cur][nb + 1 * 16 * BK];
        v16bf b2 = *(const v16bf*)&lds_b[cur][nb + 2 * 16 * BK];
        v16bf b3 = *(const v16bf*)&lds_b[cur][nb + 3 * 16 * BK];

        acc0 = __builtin_amdgcn_wmma_f32_16x16x32_bf16(
            false, afrag, false, b0, (short)0, acc0, false, false);
        acc1 = __builtin_amdgcn_wmma_f32_16x16x32_bf16(
            false, afrag, false, b1, (short)0, acc1, false, false);
        acc2 = __builtin_amdgcn_wmma_f32_16x16x32_bf16(
            false, afrag, false, b2, (short)0, acc2, false, false);
        acc3 = __builtin_amdgcn_wmma_f32_16x16x32_bf16(
            false, afrag, false, b3, (short)0, acc3, false, false);

#if USE_TDM
        if (wave == 0) TENSOR_WAIT();
#else
        ASYNC_WAIT();
#endif
        __syncthreads();
    }

    // ---- epilogue: masked bias + fp32 store ----
    // C/D layout: VGPR r -> row = sel*8 + r, col = l15
    const int obase = M0 + mw * 16 + sel * 8;
    float bb[8];
#pragma unroll
    for (int r = 0; r < 8; ++r) {
        int o = obase + r;
        bb[r] = (ub[b * COUT + o] > PDROP) ? bias[o] : 0.0f;
    }
#pragma unroll
    for (int r = 0; r < 8; ++r) {
        int o = obase + r;
        float* yr = y + ((size_t)(b * COUT + o) << 12) + N0 + nw * 64 + l15;
        yr[0]  = acc0[r] + bb[r];
        yr[16] = acc1[r] + bb[r];
        yr[32] = acc2[r] + bb[r];
        yr[48] = acc3[r] + bb[r];
    }
}

// ---------------------------------------------------------------------------
extern "C" void kernel_launch(void* const* d_in, const int* in_sizes, int n_in,
                              void* d_out, int out_size, void* d_ws, size_t ws_size,
                              hipStream_t stream)
{
    const float* x    = (const float*)d_in[0];
    const float* wgt  = (const float*)d_in[1];
    const float* bias = (const float*)d_in[2];
    const float* u_w  = (const float*)d_in[3];
    const float* u_b  = (const float*)d_in[4];
    float* y   = (float*)d_out;
    __bf16* wb = (__bf16*)d_ws;   // 32*9*256*128 bf16 = 18.9 MB scratch

    const int total4 = (B_ * NTAP * COUT * CIN) / 4;   // 2,359,296
    prep_weights<<<total4 / 256, 256, 0, stream>>>(wgt, u_w, wb);

    dim3 grid(NPIX / BN, COUT / BM, B_);               // (32, 4, 32)
    conv_wmma<<<grid, 256, 0, stream>>>(x, wb, bias, u_b, y);
}
